// ELANSwin_90606630076853
// MI455X (gfx1250) — matrix operand order repaired and checked by reading, not compiled, not run on hardware
//
#include <hip/hip_runtime.h>

typedef __attribute__((ext_vector_type(16))) _Float16 v16h;
typedef __attribute__((ext_vector_type(8)))  _Float16 v8h;
typedef __attribute__((ext_vector_type(8)))  float    v8f;
typedef __attribute__((vector_size(16)))     int      v4i;

#define HWPIX 16384
#define WIMG  128

#if __has_builtin(__builtin_amdgcn_global_load_async_to_lds_b128) && \
    __has_builtin(__builtin_amdgcn_s_wait_asynccnt)
#define HAVE_ASYNC_LDS 1
#else
#define HAVE_ASYNC_LDS 0
#endif

__device__ __forceinline__ v8f vzero8f() { v8f z = {}; return z; }
__device__ __forceinline__ v8h vzero8h() { v8h z = {}; return z; }

#if HAVE_ASYNC_LDS
typedef __attribute__((address_space(1))) v4i gv4i;  // global int4
typedef __attribute__((address_space(3))) v4i lv4i;  // LDS int4
// DMA 16 bytes global -> LDS (tracked by ASYNCcnt, no VGPR round-trip)
__device__ __forceinline__ void async_cp16(const void* g, void* l) {
  __builtin_amdgcn_global_load_async_to_lds_b128((gv4i*)(void*)g, (lv4i*)l, 0, 0);
}
__device__ __forceinline__ void async_wait0() {
  __builtin_amdgcn_s_wait_asynccnt(0);
}
#endif

__device__ __forceinline__ v8f wmma32(v16h a, v16h b, v8f c) {
  return __builtin_amdgcn_wmma_f32_16x16x32_f16(false, a, false, b, (short)0, c, false, false);
}

// A fragment (16x32 f16): lane L -> row m0+(L&15), K halves {8h..8h+7} and {16+8h..}
__device__ __forceinline__ v16h load_A(const _Float16* A, int lda, int m0, int k0, int lane) {
  int m = m0 + (lane & 15);
  int kb = k0 + ((lane >> 4) << 3);
  const _Float16* p = A + m * lda + kb;
  v16h f;
  *(v8h*)&f       = *(const v8h*)p;
  *((v8h*)&f + 1) = *(const v8h*)(p + 16);
  return f;
}

// B fragment (32x16) from B^T storage [N][K] (== torch-style [out][in] weights):
// lane L -> col n0+(L&15), K halves {8h..8h+7} and {16+8h..}
__device__ __forceinline__ v16h load_Bt(const _Float16* Bt, int ldb, int n0, int k0, int lane) {
  int n = n0 + (lane & 15);
  int kb = k0 + ((lane >> 4) << 3);
  const _Float16* p = Bt + n * ldb + kb;
  v16h f;
  *(v8h*)&f       = *(const v8h*)p;
  *((v8h*)&f + 1) = *(const v8h*)(p + 16);
  return f;
}

__device__ __forceinline__ float siluf(float x) { return x / (1.f + __expf(-x)); }

// ---------------- prep kernels ----------------
__global__ void k_f32_to_f16(const float* in, _Float16* out, int n) {
  int i = blockIdx.x * 256 + threadIdx.x;
  if (i < n) out[i] = (_Float16)in[i];
}

// OIHW [4][64][64][3][3] -> [4][64 out][tap*64+ci] with tap=ky*3+kx
__global__ void k_wblk(const float* w, _Float16* out) {
  int g = blockIdx.x * 256 + threadIdx.x;
  if (g >= 4 * 64 * 576) return;
  int i = g / (64 * 576); int rem = g % (64 * 576);
  int o = rem / 576; int kk = rem % 576;
  int tap = kk >> 6, ci = kk & 63;
  out[g] = (_Float16)w[(((i * 64 + o) * 64) + ci) * 9 + tap];
}

// continuous position bias MLP on 15x15 table -> blog[225][8]
__global__ void k_cpb(const float* w1, const float* b1, const float* w2, float* blog) {
  int t = threadIdx.x;
  if (t >= 225) return;
  int a = t / 15, bb = t % 15;
  float u0 = (float)(a - 7) * (8.0f / 7.0f);
  float u1 = (float)(bb - 7) * (8.0f / 7.0f);
  float t0 = (u0 >= 0.f ? 1.f : -1.f) * log2f(fabsf(u0) + 1.f) * (1.f / 3.f);
  float t1 = (u1 >= 0.f ? 1.f : -1.f) * log2f(fabsf(u1) + 1.f) * (1.f / 3.f);
  float acc[8] = {0.f, 0.f, 0.f, 0.f, 0.f, 0.f, 0.f, 0.f};
  for (int j = 0; j < 512; ++j) {
    float h = t0 * w1[2 * j] + t1 * w1[2 * j + 1] + b1[j];
    h = fmaxf(h, 0.f);
    for (int hd = 0; hd < 8; ++hd) acc[hd] += h * w2[hd * 512 + j];
  }
  for (int hd = 0; hd < 8; ++hd) blog[t * 8 + hd] = acc[hd];
}

// expand to bias[head][query 64][key 64] with 16*sigmoid
__global__ void k_bias(const float* blog, float* bias) {
  int g = blockIdx.x * 256 + threadIdx.x;
  if (g >= 8 * 4096) return;
  int hd = g >> 12; int rem = g & 4095;
  int n = rem >> 6, m = rem & 63;  // n=query token, m=key token
  int dr = (n >> 3) - (m >> 3) + 7;
  int dc = (n & 7) - (m & 7) + 7;
  float v = blog[(dr * 15 + dc) * 8 + hd];
  bias[g] = 16.f / (1.f + __expf(-v));
}

// ---------------- stem: 1x1 conv (both branches fused, out ch 0..127 -> e cols 128..255)
__global__ void __launch_bounds__(128) k_stem(const float* __restrict__ x,
    const _Float16* __restrict__ wst, const float* __restrict__ sst,
    const float* __restrict__ bst, _Float16* __restrict__ e) {
  __shared__ _Float16 ldsA[64 * 128];
  int t = threadIdx.x, lane = t & 31, wv = t >> 5;
  int p0 = blockIdx.x * 64;
  int b = p0 >> 14, hw0 = p0 & (HWPIX - 1);
  const float* base = x + (size_t)b * 128 * HWPIX + hw0;
  for (int i = t; i < 64 * 128; i += 128) {
    int c = i >> 6, m = i & 63;
    ldsA[m * 128 + c] = (_Float16)base[(size_t)c * HWPIX + m];
  }
  __syncthreads();
  int m0 = wv * 16;
  v8f acc[8];
  for (int j = 0; j < 8; ++j) acc[j] = vzero8f();
  for (int kt = 0; kt < 4; ++kt) {
    v16h a = load_A(ldsA, 128, m0, kt * 32, lane);
    for (int j = 0; j < 8; ++j)
      acc[j] = wmma32(a, load_Bt(wst, 128, j * 16, kt * 32, lane), acc[j]);
  }
  for (int j = 0; j < 8; ++j) {
    int o = j * 16 + (lane & 15);
    float sc = sst[o], bi = bst[o];
    for (int r = 0; r < 8; ++r) {
      int m = m0 + r + ((lane >> 4) << 3);
      float y = siluf(acc[j][r] * sc + bi);
      e[(size_t)(p0 + m) * 256 + 128 + o] = (_Float16)y;
    }
  }
}

// ---------------- 3x3 conv as implicit GEMM (K = 9*64 = 576), 8x8 output tile per WG
__global__ void __launch_bounds__(128) k_conv3(const _Float16* __restrict__ in, int in_stride,
    _Float16* __restrict__ out, int out_stride, const _Float16* __restrict__ wt,
    const float* __restrict__ sc, const float* __restrict__ bi) {
  __shared__ _Float16 patch[100 * 64];  // 10x10 halo x 64ch
  int t = threadIdx.x, lane = t & 31, wv = t >> 5;
  int tid = blockIdx.x;
  int b = tid >> 8; int rem = tid & 255;
  int y0 = (rem >> 4) << 3, x0 = (rem & 15) << 3;
  for (int pp = t; pp < 100; pp += 128) {
    int py = y0 + pp / 10 - 1, px = x0 + pp % 10 - 1;
    v8h* dst = (v8h*)(patch + pp * 64);
    if (py >= 0 && py < 128 && px >= 0 && px < 128) {
      const v8h* src = (const v8h*)(in + (size_t)(b * HWPIX + py * WIMG + px) * in_stride);
#if HAVE_ASYNC_LDS
      for (int j = 0; j < 8; ++j) async_cp16(src + j, dst + j);
#else
      for (int j = 0; j < 8; ++j) dst[j] = src[j];
#endif
    } else {
      v8h z = vzero8h();
      for (int j = 0; j < 8; ++j) dst[j] = z;
    }
  }
#if HAVE_ASYNC_LDS
  async_wait0();
#endif
  __syncthreads();
  int m0 = wv * 16;
  int m = m0 + (lane & 15);
  int my = m >> 3, mx = m & 7;
  v8f acc[4];
  for (int j = 0; j < 4; ++j) acc[j] = vzero8f();
  for (int kt = 0; kt < 18; ++kt) {
    int kb = kt * 32 + ((lane >> 4) << 3);
    v16h a;
    {
      int tap = kb >> 6, ci = kb & 63;
      int pr = (my + tap / 3) * 10 + (mx + tap % 3);
      *(v8h*)&a = *(const v8h*)(patch + pr * 64 + ci);
      int kb2 = kb + 16;
      tap = kb2 >> 6; ci = kb2 & 63;
      pr = (my + tap / 3) * 10 + (mx + tap % 3);
      *((v8h*)&a + 1) = *(const v8h*)(patch + pr * 64 + ci);
    }
    for (int j = 0; j < 4; ++j)
      acc[j] = wmma32(a, load_Bt(wt, 576, j * 16, kt * 32, lane), acc[j]);
  }
  for (int j = 0; j < 4; ++j) {
    int o = j * 16 + (lane & 15);
    float s = sc[o], bb = bi[o];
    for (int r = 0; r < 8; ++r) {
      int mm = m0 + r + ((lane >> 4) << 3);
      int pix = b * HWPIX + (y0 + (mm >> 3)) * WIMG + x0 + (mm & 7);
      float y = siluf(acc[j][r] * s + bb);
      out[(size_t)pix * out_stride + o] = (_Float16)y;
    }
  }
}

// ---------------- fused window attention: QKV -> cos-attn -> softmax -> PV -> proj -> LN+res
__global__ void __launch_bounds__(128) k_attn(const _Float16* __restrict__ e,
    const _Float16* __restrict__ wqkv, const float* __restrict__ qkvb,
    const _Float16* __restrict__ wproj, const float* __restrict__ projb,
    const float* __restrict__ lsc, const float* __restrict__ bias,
    const float* __restrict__ n1g, const float* __restrict__ n1b,
    _Float16* __restrict__ sbuf) {
  __shared__ __align__(32) char smem[159808];
  _Float16* qkvh = (_Float16*)smem;            // [64][512]   q|k (head-major cols)
  _Float16* vt   = (_Float16*)(smem + 65536);  // [256][64]   V transposed: [h*32+d][token]
  _Float16* Oh   = (_Float16*)(smem + 98304);  // [64][256]
  float*    Sc   = (float*)(smem + 131072);    // [64][64] scores (per head)
  _Float16* Ph   = (_Float16*)(smem + 147456); // [64][64] probs
  float*    rn   = (float*)(smem + 155648);    // [64][16] 1/max(norm,eps): q h0-7, k h8-15
  float*    scl  = (float*)(smem + 159744);    // [8]
  float*    Pr   = (float*)smem;               // overlay [64][256] f32 (after heads done)

  int t = threadIdx.x, lane = t & 31, wv = t >> 5;
  int wid = blockIdx.x;
  int b = wid >> 8, rem = wid & 255, wy = rem >> 4, wx = rem & 15;
  int m0 = wv * 16;
  int ml = m0 + (lane & 15);
  int pixl = b * HWPIX + (wy * 8 + (ml >> 3)) * WIMG + wx * 8 + (ml & 7);
  const _Float16* arow = e + (size_t)pixl * 256;
  if (t < 8) scl[t] = __expf(fminf(lsc[t], 4.60517019f));  // exp(min(ls, log 100))

  // QKV: [64,256] @ [256,768]
  for (int ch = 0; ch < 6; ++ch) {
    v8f acc[8];
    for (int j = 0; j < 8; ++j) acc[j] = vzero8f();
    for (int kt = 0; kt < 8; ++kt) {
      int kb = kt * 32 + ((lane >> 4) << 3);
      v16h a;
      *(v8h*)&a       = *(const v8h*)(arow + kb);
      *((v8h*)&a + 1) = *(const v8h*)(arow + kb + 16);
      for (int j = 0; j < 8; ++j)
        acc[j] = wmma32(a, load_Bt(wqkv, 256, (ch * 8 + j) * 16, kt * 32, lane), acc[j]);
    }
    for (int j = 0; j < 8; ++j) {
      int col = (ch * 8 + j) * 16 + (lane & 15);
      float bb = qkvb[col];
      for (int r = 0; r < 8; ++r) {
        int m = m0 + r + ((lane >> 4) << 3);
        float y = acc[j][r] + bb;
        if (col < 512) qkvh[m * 512 + col] = (_Float16)y;
        else           vt[(col - 512) * 64 + m] = (_Float16)y;
      }
    }
  }
  __syncthreads();
  // reciprocal row norms for q and k
  for (int pr = t; pr < 64 * 16; pr += 128) {
    int m = pr >> 4, h = pr & 15;
    const _Float16* bp = qkvh + m * 512 + ((h < 8) ? h * 32 : 256 + (h - 8) * 32);
    float s = 0.f;
    for (int d = 0; d < 32; ++d) { float v = (float)bp[d]; s += v * v; }
    rn[pr] = 1.f / fmaxf(sqrtf(s), 1e-12f);
  }
  __syncthreads();

  for (int h = 0; h < 8; ++h) {
    // scores S[m][n] = scale * q.k / (|q||k|) + bias
    v16h aq = load_A(qkvh, 512, m0, h * 32, lane);
    float sch = scl[h];
    for (int nt = 0; nt < 4; ++nt) {
      v8f acc = wmma32(aq, load_Bt(qkvh, 512, nt * 16, 256 + h * 32, lane), vzero8f());
      int n = nt * 16 + (lane & 15);
      float rk = rn[n * 16 + 8 + h];
      for (int r = 0; r < 8; ++r) {
        int m = m0 + r + ((lane >> 4) << 3);
        Sc[m * 64 + n] = acc[r] * sch * rn[m * 16 + h] * rk + bias[h * 4096 + m * 64 + n];
      }
    }
    __syncthreads();
    if (t < 64) {  // softmax over keys
      float mx = -1e30f;
      for (int n = 0; n < 64; ++n) mx = fmaxf(mx, Sc[t * 64 + n]);
      float sum = 0.f;
      for (int n = 0; n < 64; ++n) sum += __expf(Sc[t * 64 + n] - mx);
      float inv = 1.f / sum;
      for (int n = 0; n < 64; ++n)
        Ph[t * 64 + n] = (_Float16)(__expf(Sc[t * 64 + n] - mx) * inv);
    }
    __syncthreads();
    // O_h = P @ V_h  ([64,64]@[64,32])
    v8f accp[2];
    accp[0] = vzero8f(); accp[1] = vzero8f();
    for (int kt = 0; kt < 2; ++kt) {
      v16h ap = load_A(Ph, 64, m0, kt * 32, lane);
      for (int nt = 0; nt < 2; ++nt)
        accp[nt] = wmma32(ap, load_Bt(vt + h * 32 * 64, 64, nt * 16, kt * 32, lane), accp[nt]);
    }
    for (int nt = 0; nt < 2; ++nt) {
      int d = nt * 16 + (lane & 15);
      for (int r = 0; r < 8; ++r) {
        int m = m0 + r + ((lane >> 4) << 3);
        Oh[m * 256 + h * 32 + d] = (_Float16)accp[nt][r];
      }
    }
    __syncthreads();
  }
  // proj: [64,256] @ [256,256] -> Pr (overlays dead qkvh region)
  for (int ch = 0; ch < 2; ++ch) {
    v8f acc[8];
    for (int j = 0; j < 8; ++j) acc[j] = vzero8f();
    for (int kt = 0; kt < 8; ++kt) {
      v16h a = load_A(Oh, 256, m0, kt * 32, lane);
      for (int j = 0; j < 8; ++j)
        acc[j] = wmma32(a, load_Bt(wproj, 256, (ch * 8 + j) * 16, kt * 32, lane), acc[j]);
    }
    for (int j = 0; j < 8; ++j) {
      int col = (ch * 8 + j) * 16 + (lane & 15);
      float bb = projb[col];
      for (int r = 0; r < 8; ++r) {
        int m = m0 + r + ((lane >> 4) << 3);
        Pr[m * 256 + col] = acc[j][r] + bb;
      }
    }
  }
  __syncthreads();
  // s = e + LayerNorm(attn_out)
  if (t < 64) {
    int m = t;
    int pix = b * HWPIX + (wy * 8 + (m >> 3)) * WIMG + wx * 8 + (m & 7);
    const _Float16* erow = e + (size_t)pix * 256;
    float su = 0.f, sq = 0.f;
    for (int c = 0; c < 256; ++c) { float v = Pr[m * 256 + c]; su += v; sq += v * v; }
    float mean = su * (1.f / 256.f);
    float var = sq * (1.f / 256.f) - mean * mean;
    float rstd = rsqrtf(var + 1e-5f);
    _Float16* srow = sbuf + (size_t)pix * 256;
    for (int c = 0; c < 256; ++c) {
      float v = (Pr[m * 256 + c] - mean) * rstd * n1g[c] + n1b[c];
      srow[c] = (_Float16)((float)erow[c] + v);
    }
  }
}

// ---------------- fused MLP (two GEMMs + GELU) + LN + residual + NCHW writeout
__global__ void __launch_bounds__(128) k_mlp(const _Float16* __restrict__ sbuf,
    const _Float16* __restrict__ w1, const float* __restrict__ b1,
    const _Float16* __restrict__ w2, const float* __restrict__ b2,
    const float* __restrict__ n2g, const float* __restrict__ n2b,
    float* __restrict__ out) {
  __shared__ __align__(32) char smem[131072];
  _Float16* sA = (_Float16*)smem;             // [64][256]
  _Float16* hA = (_Float16*)(smem + 32768);   // [64][256]
  float*    h2 = (float*)(smem + 65536);      // [64][256]
  int t = threadIdx.x, lane = t & 31, wv = t >> 5;
  int p0 = blockIdx.x * 64;
  const v8h* gs = (const v8h*)(sbuf + (size_t)p0 * 256);
  v8h* ls = (v8h*)sA;
#if HAVE_ASYNC_LDS
  for (int i = t; i < 2048; i += 128) async_cp16(gs + i, ls + i);
  async_wait0();
#else
  for (int i = t; i < 2048; i += 128) {
    __builtin_prefetch(gs + i + 512, 0, 3);
    ls[i] = gs[i];
  }
#endif
  __syncthreads();
  int m0 = wv * 16;
  // GEMM1 + GELU(exact)
  for (int ch = 0; ch < 2; ++ch) {
    v8f acc[8];
    for (int j = 0; j < 8; ++j) acc[j] = vzero8f();
    for (int kt = 0; kt < 8; ++kt) {
      v16h a = load_A(sA, 256, m0, kt * 32, lane);
      for (int j = 0; j < 8; ++j)
        acc[j] = wmma32(a, load_Bt(w1, 256, (ch * 8 + j) * 16, kt * 32, lane), acc[j]);
    }
    for (int j = 0; j < 8; ++j) {
      int col = (ch * 8 + j) * 16 + (lane & 15);
      float bb = b1[col];
      for (int r = 0; r < 8; ++r) {
        int m = m0 + r + ((lane >> 4) << 3);
        float y = acc[j][r] + bb;
        y = 0.5f * y * (1.f + erff(y * 0.70710678f));
        hA[m * 256 + col] = (_Float16)y;
      }
    }
  }
  __syncthreads();
  // GEMM2
  for (int ch = 0; ch < 2; ++ch) {
    v8f acc[8];
    for (int j = 0; j < 8; ++j) acc[j] = vzero8f();
    for (int kt = 0; kt < 8; ++kt) {
      v16h a = load_A(hA, 256, m0, kt * 32, lane);
      for (int j = 0; j < 8; ++j)
        acc[j] = wmma32(a, load_Bt(w2, 256, (ch * 8 + j) * 16, kt * 32, lane), acc[j]);
    }
    for (int j = 0; j < 8; ++j) {
      int col = (ch * 8 + j) * 16 + (lane & 15);
      float bb = b2[col];
      for (int r = 0; r < 8; ++r) {
        int m = m0 + r + ((lane >> 4) << 3);
        h2[m * 256 + col] = acc[j][r] + bb;
      }
    }
  }
  __syncthreads();
  // out = s + LayerNorm(h2), in place
  if (t < 64) {
    int m = t;
    float su = 0.f, sq = 0.f;
    for (int c = 0; c < 256; ++c) { float v = h2[m * 256 + c]; su += v; sq += v * v; }
    float mean = su * (1.f / 256.f);
    float var = sq * (1.f / 256.f) - mean * mean;
    float rstd = rsqrtf(var + 1e-5f);
    for (int c = 0; c < 256; ++c) {
      float v = (h2[m * 256 + c] - mean) * rstd * n2g[c] + n2b[c];
      h2[m * 256 + c] = (float)sA[m * 256 + c] + v;
    }
  }
  __syncthreads();
  // coalesced NCHW writeout
  int b = p0 >> 14, hw0 = p0 & (HWPIX - 1);
  float* ob = out + (size_t)b * 256 * HWPIX + hw0;
  int m = t & 63;
  for (int cc = 0; cc < 128; ++cc) {
    int c = cc * 2 + (t >> 6);
    ob[(size_t)c * HWPIX + m] = h2[m * 256 + c];
  }
}

// ---------------- host ----------------
extern "C" void kernel_launch(void* const* d_in, const int* in_sizes, int n_in,
                              void* d_out, int out_size, void* d_ws, size_t ws_size,
                              hipStream_t stream) {
  const float* x      = (const float*)d_in[0];
  const float* w_stem = (const float*)d_in[1];
  const float* s_stem = (const float*)d_in[2];
  const float* b_stem = (const float*)d_in[3];
  const float* w_blk  = (const float*)d_in[4];
  const float* s_blk  = (const float*)d_in[5];
  const float* b_blk  = (const float*)d_in[6];
  const float* n1g    = (const float*)d_in[7];
  const float* n1b    = (const float*)d_in[8];
  const float* qkv_w  = (const float*)d_in[9];
  const float* qkv_b  = (const float*)d_in[10];
  const float* proj_w = (const float*)d_in[11];
  const float* proj_b = (const float*)d_in[12];
  const float* lsc    = (const float*)d_in[13];
  const float* cpb_w1 = (const float*)d_in[14];
  const float* cpb_b1 = (const float*)d_in[15];
  const float* cpb_w2 = (const float*)d_in[16];
  const float* n2g    = (const float*)d_in[17];
  const float* n2b    = (const float*)d_in[18];
  const float* mlp_w1 = (const float*)d_in[19];
  const float* mlp_b1 = (const float*)d_in[20];
  const float* mlp_w2 = (const float*)d_in[21];
  const float* mlp_b2 = (const float*)d_in[22];
  float* out = (float*)d_out;

  char* ws = (char*)d_ws;
  _Float16* e     = (_Float16*)(ws + 0);          // [65536][256] NHWC f16
  _Float16* t0    = (_Float16*)(ws + 33554432);   // [65536][64] f16
  _Float16* sbuf  = (_Float16*)(ws + 41943040);   // [65536][256] f16
  _Float16* wstem = (_Float16*)(ws + 75497472);   // [128][128]
  _Float16* wblk  = (_Float16*)(ws + 75530240);   // [4][64][576]
  _Float16* wqkv  = (_Float16*)(ws + 75825152);   // [768][256]
  _Float16* wproj = (_Float16*)(ws + 76218368);   // [256][256]
  _Float16* wm1   = (_Float16*)(ws + 76349440);   // [256][256]
  _Float16* wm2   = (_Float16*)(ws + 76480512);   // [256][256]
  float*    bias  = (float*)(ws + 76611584);      // [8][64][64]
  float*    blog  = (float*)(ws + 76742656);      // [225][8]

  (void)in_sizes; (void)n_in; (void)out_size; (void)ws_size;

  k_f32_to_f16<<<(16384 + 255) / 256, 256, 0, stream>>>(w_stem, wstem, 16384);
  k_f32_to_f16<<<(196608 + 255) / 256, 256, 0, stream>>>(qkv_w, wqkv, 196608);
  k_f32_to_f16<<<(65536 + 255) / 256, 256, 0, stream>>>(proj_w, wproj, 65536);
  k_f32_to_f16<<<(65536 + 255) / 256, 256, 0, stream>>>(mlp_w1, wm1, 65536);
  k_f32_to_f16<<<(65536 + 255) / 256, 256, 0, stream>>>(mlp_w2, wm2, 65536);
  k_wblk<<<(4 * 64 * 576 + 255) / 256, 256, 0, stream>>>(w_blk, wblk);
  k_cpb<<<1, 256, 0, stream>>>(cpb_w1, cpb_b1, cpb_w2, blog);
  k_bias<<<(8 * 4096) / 256, 256, 0, stream>>>(blog, bias);

  k_stem<<<1024, 128, 0, stream>>>(x, wstem, s_stem, b_stem, e);
  // ELAN chain: main(e+128) -> t0 -> outs[0](e+64) -> t0 -> outs[1](e+0)
  k_conv3<<<1024, 128, 0, stream>>>(e + 128, 256, t0, 64, wblk + 0 * 36864, s_blk + 0, b_blk + 0);
  k_conv3<<<1024, 128, 0, stream>>>(t0, 64, e + 64, 256, wblk + 1 * 36864, s_blk + 64, b_blk + 64);
  k_conv3<<<1024, 128, 0, stream>>>(e + 64, 256, t0, 64, wblk + 2 * 36864, s_blk + 128, b_blk + 128);
  k_conv3<<<1024, 128, 0, stream>>>(t0, 64, e + 0, 256, wblk + 3 * 36864, s_blk + 192, b_blk + 192);

  k_attn<<<1024, 128, 0, stream>>>(e, wqkv, qkv_b, wproj, proj_b, lsc, bias, n1g, n1b, sbuf);
  k_mlp<<<1024, 128, 0, stream>>>(sbuf, wm1, mlp_b1, wm2, mlp_b2, n2g, n2b, out);
}
